// RNN_11115375362882
// MI455X (gfx1250) — matrix operand order repaired and checked by reading, not compiled, not run on hardware
//
#include <hip/hip_runtime.h>
#include <math.h>

// ---------------------------------------------------------------------------
// Problem constants (from reference): S=512, B=64, V=50000, E=H=512
// ---------------------------------------------------------------------------
#define S_LEN 512
#define BATCH 64
#define HID   512
#define ROWS  (S_LEN * BATCH)   // 32768 rows of the big input-projection GEMMs

typedef __attribute__((ext_vector_type(16))) __bf16 v16bf;
typedef __attribute__((ext_vector_type(8)))  float  v8f;

union Frag {
    v16bf  v;
    uint4  q[2];
    __bf16 h[16];
};

__device__ __forceinline__ v8f wmma_bf16(const Frag& a, const Frag& b, v8f c) {
    // D(f32 16x16) = A(16x32 bf16) * B(32x16 bf16) + C
    return __builtin_amdgcn_wmma_f32_16x16x32_bf16(
        /*neg_a=*/false, a.v, /*neg_b=*/false, b.v,
        /*c_mod=*/(short)0, c, /*reuse_a=*/false, /*reuse_b=*/false);
}

// Branchless tanh: 1 - 2/(exp(2x)+1). Saturates exactly at +-1 for large |x|;
// abs error ~1e-6 which is far below the bf16 storage quantization of h.
__device__ __forceinline__ float fast_tanh(float x) {
    float e = __expf(2.0f * x);                 // v_exp_f32 path
    float r = __builtin_amdgcn_rcpf(e + 1.0f);  // v_rcp_f32
    return 1.0f - 2.0f * r;                     // fma
}

// B fragment: column n of B == row n of W (row-major, contiguous in k).
// Lanes 0-15 hold K {kb..kb+7, kb+16..kb+23}; lanes 16-31: shifted by 8.
__device__ __forceinline__ void load_b_frag(Frag& b, const __bf16* __restrict__ W,
                                            int n, int kb, bool lo) {
    const __bf16* brow = W + (size_t)n * HID;
    const int bo = kb + (lo ? 0 : 8);
    b.q[0] = *(const uint4*)(brow + bo);
    b.q[1] = *(const uint4*)(brow + bo + 16);
}

// A fragment from a bf16 row (global or LDS). Lane<16: K {kb..kb+7, kb+16..kb+23};
// lane>=16: shifted by 8.
__device__ __forceinline__ void load_a_bf16(Frag& a, const __bf16* arow,
                                            int kb, bool lo) {
    const int ko = kb + (lo ? 0 : 8);
    a.q[0] = *(const uint4*)(arow + ko);
    a.q[1] = *(const uint4*)(arow + ko + 16);
}

// A fragment gathered from an f32 row, converted to bf16 in registers.
__device__ __forceinline__ void load_a_f32(Frag& a, const float* __restrict__ arow,
                                           int kb, bool lo) {
    const int ko = kb + (lo ? 0 : 8);
    float af[16];
    *(float4*)&af[0]  = *(const float4*)(arow + ko);
    *(float4*)&af[4]  = *(const float4*)(arow + ko + 4);
    *(float4*)&af[8]  = *(const float4*)(arow + ko + 16);
    *(float4*)&af[12] = *(const float4*)(arow + ko + 20);
    #pragma unroll
    for (int i = 0; i < 16; ++i) a.h[i] = (__bf16)af[i];
}

// ---------------------------------------------------------------------------
// Kernel 1: f32 -> bf16 weight conversion, 8 elements/thread
// ---------------------------------------------------------------------------
__global__ void cvt_f32_bf16_v8(const float* __restrict__ src,
                                __bf16* __restrict__ dst, int n8) {
    int i = blockIdx.x * blockDim.x + threadIdx.x;
    if (i >= n8) return;
    float4 f0 = ((const float4*)src)[2 * i];
    float4 f1 = ((const float4*)src)[2 * i + 1];
    union { __bf16 h[8]; uint4 q; } u;
    u.h[0] = (__bf16)f0.x; u.h[1] = (__bf16)f0.y;
    u.h[2] = (__bf16)f0.z; u.h[3] = (__bf16)f0.w;
    u.h[4] = (__bf16)f1.x; u.h[5] = (__bf16)f1.y;
    u.h[6] = (__bf16)f1.z; u.h[7] = (__bf16)f1.w;
    ((uint4*)dst)[i] = u.q;
}

// ---------------------------------------------------------------------------
// Kernel 2: xproj[r][n] = sum_k A[r][k] * W[n][k] + b1[n] + b2[n]
//   MODE 0: A row r = emb[idx[r]]  (f32 gather, converted to bf16 in regs)
//   MODE 1: A row r = Abf[r]       (bf16, already materialized)
// One wave computes a 16(M) x 64(N) stripe; 8 waves / block; 256 blocks.
// 2-stage software pipeline over K so VMEM latency hides behind WMMAs.
// ---------------------------------------------------------------------------
template <int MODE>
__global__ void __launch_bounds__(256)
xproj_gemm(const float* __restrict__ emb, const int* __restrict__ idx,
           const __bf16* __restrict__ Abf, const __bf16* __restrict__ Wbf,
           const float* __restrict__ b1, const float* __restrict__ b2,
           float* __restrict__ xproj) {
    const int lane    = threadIdx.x & 31;
    const int wave    = threadIdx.x >> 5;
    const int gw      = blockIdx.x * 8 + wave;
    const int m_tile  = gw >> 3;       // 0..255 (16 rows each)
    const int n_group = gw & 7;        // 0..7   (64 cols each)
    const bool lo     = (lane < 16);
    const int  col    = lane & 15;
    const int  m      = m_tile * 16 + col;   // A-matrix row owned by this lane

    const float*  arow_f = (MODE == 0) ? (emb + (size_t)idx[m] * HID) : nullptr;
    const __bf16* arow_b = (MODE == 1) ? (Abf + (size_t)m * HID)      : nullptr;

    v8f acc[4];
    #pragma unroll
    for (int nt = 0; nt < 4; ++nt) {
        const int n = n_group * 64 + nt * 16 + col;
        const float bias = b1[n] + b2[n];   // C col depends only on lane -> broadcast
        #pragma unroll
        for (int r = 0; r < 8; ++r) acc[nt][r] = bias;
    }

    Frag a0, a1, b0[4], b1f[4];
    if (MODE == 0) load_a_f32(a0, arow_f, 0, lo); else load_a_bf16(a0, arow_b, 0, lo);
    #pragma unroll
    for (int nt = 0; nt < 4; ++nt)
        load_b_frag(b0[nt], Wbf, n_group * 64 + nt * 16 + col, 0, lo);

    #pragma unroll 1
    for (int kb = 0; kb < HID; kb += 64) {
        // prefetch kb+32, compute kb
        if (MODE == 0) load_a_f32(a1, arow_f, kb + 32, lo);
        else           load_a_bf16(a1, arow_b, kb + 32, lo);
        #pragma unroll
        for (int nt = 0; nt < 4; ++nt)
            load_b_frag(b1f[nt], Wbf, n_group * 64 + nt * 16 + col, kb + 32, lo);
        #pragma unroll
        for (int nt = 0; nt < 4; ++nt) acc[nt] = wmma_bf16(a0, b0[nt], acc[nt]);

        // prefetch kb+64 (if any), compute kb+32
        if (kb + 64 < HID) {
            if (MODE == 0) load_a_f32(a0, arow_f, kb + 64, lo);
            else           load_a_bf16(a0, arow_b, kb + 64, lo);
            #pragma unroll
            for (int nt = 0; nt < 4; ++nt)
                load_b_frag(b0[nt], Wbf, n_group * 64 + nt * 16 + col, kb + 64, lo);
        }
        #pragma unroll
        for (int nt = 0; nt < 4; ++nt) acc[nt] = wmma_bf16(a1, b1f[nt], acc[nt]);
    }

    #pragma unroll
    for (int nt = 0; nt < 4; ++nt) {
        const int n = n_group * 64 + nt * 16 + col;
        #pragma unroll
        for (int r = 0; r < 8; ++r) {
            const int row = m_tile * 16 + r + (lo ? 0 : 8);
            xproj[(size_t)row * HID + n] = acc[nt][r];
        }
    }
}

// ---------------------------------------------------------------------------
// Kernel 3: persistent single-WGP recurrent scan.
//   h_t = tanh(xproj_t + h_{t-1} @ Whh^T),  h in LDS (64x512 bf16 = 64KB).
//   1024 threads = 32 wave32s (8 per SIMD): wave owns m_tile = w&3 (16 rows)
//   x n_group = w>>2 (64 cols, 4 WMMA tiles). Low register pressure per wave
//   (acc 32 + A 8 + B 32 VGPRs); deep wave occupancy hides L2 latency.
// ---------------------------------------------------------------------------
__global__ void __launch_bounds__(1024, 1)
rnn_scan(const float* __restrict__ xproj, const __bf16* __restrict__ Whhbf,
         __bf16* __restrict__ hseq, float* __restrict__ hlast, int steps) {
    __shared__ __bf16 hcur[BATCH * HID];   // 64 KB of the WGP's 320 KB LDS

    const int lane    = threadIdx.x & 31;
    const int wave    = threadIdx.x >> 5;   // 0..31
    const int m_tile  = wave & 3;           // 4 tiles cover M=64
    const int n_group = wave >> 2;          // 8 groups x 64 cols cover N=512
    const bool lo     = (lane < 16);
    const int  col    = lane & 15;
    const int  m      = m_tile * 16 + col;
    const bool store_seq  = (hseq != nullptr);
    const bool store_last = (hlast != nullptr);

    // h0 = 0
    {
        unsigned* p = (unsigned*)hcur;
        for (int i = threadIdx.x; i < (BATCH * HID) / 2; i += 1024) p[i] = 0u;
    }
    __syncthreads();

    for (int t = 0; t < steps; ++t) {
        const float* xp = xproj + (size_t)t * (BATCH * HID);
        if (t + 1 < steps)  // warm next step's C-init slice into cache
            __builtin_prefetch(xproj + (size_t)(t + 1) * (BATCH * HID)
                                     + threadIdx.x * 128, 0, 0);

        v8f acc[4];
        #pragma unroll
        for (int nt = 0; nt < 4; ++nt) {
            const int n = n_group * 64 + nt * 16 + col;
            #pragma unroll
            for (int r = 0; r < 8; ++r) {
                const int row = m_tile * 16 + r + (lo ? 0 : 8);
                acc[nt][r] = xp[row * HID + n];
            }
        }

        #pragma unroll 1
        for (int kb = 0; kb < HID; kb += 32) {
            Frag a;                                    // A from LDS h (bf16)
            load_a_bf16(a, &hcur[m * HID], kb, lo);
            Frag b[4];
            #pragma unroll
            for (int nt = 0; nt < 4; ++nt)
                load_b_frag(b[nt], Whhbf, n_group * 64 + nt * 16 + col, kb, lo);
            #pragma unroll
            for (int nt = 0; nt < 4; ++nt)
                acc[nt] = wmma_bf16(a, b[nt], acc[nt]);
        }

        __syncthreads();   // everyone finished reading h_{t-1} from LDS

        // branchless tanh in place (no EXEC divergence on the critical path)
        #pragma unroll
        for (int nt = 0; nt < 4; ++nt)
            #pragma unroll
            for (int r = 0; r < 8; ++r)
                acc[nt][r] = fast_tanh(acc[nt][r]);

        // write h_t back to LDS (always)
        #pragma unroll
        for (int nt = 0; nt < 4; ++nt) {
            const int n = n_group * 64 + nt * 16 + col;
            #pragma unroll
            for (int r = 0; r < 8; ++r) {
                const int row = m_tile * 16 + r + (lo ? 0 : 8);
                hcur[row * HID + n] = (__bf16)acc[nt][r];
            }
        }
        // stream h_t to global (layer 0 only)
        if (store_seq) {
            __bf16* hs = hseq + (size_t)t * (BATCH * HID);
            #pragma unroll
            for (int nt = 0; nt < 4; ++nt) {
                const int n = n_group * 64 + nt * 16 + col;
                #pragma unroll
                for (int r = 0; r < 8; ++r) {
                    const int row = m_tile * 16 + r + (lo ? 0 : 8);
                    hs[row * HID + n] = (__bf16)acc[nt][r];
                }
            }
        }
        // keep final h in f32 (layer 1 only)
        if (store_last && t == steps - 1) {
            #pragma unroll
            for (int nt = 0; nt < 4; ++nt) {
                const int n = n_group * 64 + nt * 16 + col;
                #pragma unroll
                for (int r = 0; r < 8; ++r) {
                    const int row = m_tile * 16 + r + (lo ? 0 : 8);
                    hlast[row * HID + n] = acc[nt][r];
                }
            }
        }
        __syncthreads();   // h_t visible before next iteration reads it
    }
}

// ---------------------------------------------------------------------------
// Kernel 4: out[b][j] = h_last[b] . fc_w[j] + fc_b[j]   (64x5, tiny)
// ---------------------------------------------------------------------------
__global__ void fc_kernel(const float* __restrict__ hlast,
                          const float* __restrict__ fcw,
                          const float* __restrict__ fcb,
                          float* __restrict__ out) {
    const int tid = blockIdx.x * blockDim.x + threadIdx.x;
    if (tid >= BATCH * 5) return;
    const int b = tid / 5, j = tid % 5;
    const float* hr = hlast + b * HID;
    const float* wr = fcw + j * HID;
    float s = fcb[j];
    for (int k = 0; k < HID; ++k) s += hr[k] * wr[k];
    out[tid] = s;
}

// ---------------------------------------------------------------------------
// Host-side orchestration
// ---------------------------------------------------------------------------
extern "C" void kernel_launch(void* const* d_in, const int* in_sizes, int n_in,
                              void* d_out, int out_size, void* d_ws, size_t ws_size,
                              hipStream_t stream) {
    const int*   x    = (const int*)  d_in[0];
    const float* emb  = (const float*)d_in[1];
    const float* Wih0 = (const float*)d_in[2];
    const float* Whh0 = (const float*)d_in[3];
    const float* bih0 = (const float*)d_in[4];
    const float* bhh0 = (const float*)d_in[5];
    const float* Wih1 = (const float*)d_in[6];
    const float* Whh1 = (const float*)d_in[7];
    const float* bih1 = (const float*)d_in[8];
    const float* bhh1 = (const float*)d_in[9];
    const float* fcw  = (const float*)d_in[10];
    const float* fcb  = (const float*)d_in[11];
    float* out = (float*)d_out;

    // Workspace layout (bytes):
    //   [0,2MB)        four bf16 512x512 weight matrices
    //   [2MB,66MB)     xproj f32  [32768 x 512]  (reused by both layers)
    //   [66MB,98MB)    h1seq bf16 [32768 x 512]
    //   [98MB,+128KB)  h_last f32 [64 x 512]
    char* ws = (char*)d_ws;
    const size_t WELEM = (size_t)HID * HID;          // 262144
    __bf16* Wih0b = (__bf16*)ws;
    __bf16* Whh0b = Wih0b + WELEM;
    __bf16* Wih1b = Whh0b + WELEM;
    __bf16* Whh1b = Wih1b + WELEM;
    float*  xproj = (float*) (ws + 4 * WELEM * 2);                       // 2 MB in
    __bf16* h1seq = (__bf16*)(ws + 4 * WELEM * 2 + (size_t)ROWS*HID*4);  // +64 MB
    float*  hlast = (float*) ((char*)h1seq + (size_t)ROWS*HID*2);        // +32 MB

    const int N8 = (int)(WELEM / 8);    // 32768 vec8 chunks per matrix
    cvt_f32_bf16_v8<<<N8 / 256, 256, 0, stream>>>(Wih0, Wih0b, N8);
    cvt_f32_bf16_v8<<<N8 / 256, 256, 0, stream>>>(Whh0, Whh0b, N8);
    cvt_f32_bf16_v8<<<N8 / 256, 256, 0, stream>>>(Wih1, Wih1b, N8);
    cvt_f32_bf16_v8<<<N8 / 256, 256, 0, stream>>>(Whh1, Whh1b, N8);

    // Layer 0: fused gather + input projection, then recurrent scan.
    xproj_gemm<0><<<ROWS / 16 / 8, 256, 0, stream>>>(emb, x, nullptr, Wih0b,
                                                     bih0, bhh0, xproj);
    rnn_scan<<<1, 1024, 0, stream>>>(xproj, Whh0b, h1seq, nullptr, S_LEN);

    // Layer 1: input projection of h1 sequence, then scan keeping last h.
    xproj_gemm<1><<<ROWS / 16 / 8, 256, 0, stream>>>(nullptr, nullptr, h1seq, Wih1b,
                                                     bih1, bhh1, xproj);
    rnn_scan<<<1, 1024, 0, stream>>>(xproj, Whh1b, nullptr, hlast, S_LEN);

    // Final classifier.
    fc_kernel<<<1, BATCH * 5, 0, stream>>>(hlast, fcw, fcb, out);
}
